// MoE_75393855914555
// MI455X (gfx1250) — matrix operand order repaired and checked by reading, not compiled, not run on hardware
//
#include <hip/hip_runtime.h>
#include <hip/hip_bf16.h>
#include <math.h>

// ---------------- problem constants (from reference) ----------------
constexpr int B_   = 4;
constexpr int N_   = 2048;
constexpr int D_   = 1024;
constexpr int DC_  = 256;
constexpr int E_   = 8;
constexpr int TOPK_= 2;
constexpr int HID_ = 4096;
constexpr int T_   = B_ * N_;          // 8192
constexpr int CAP_ = 2560;             // int(1.25 * 2 * 8192 / 8)
constexpr int EG_  = E_ * CAP_;        // 20480
constexpr int TD3_ = 3 * D_;           // 3072

// ---------------- types ----------------
typedef __attribute__((ext_vector_type(16))) __bf16 v16bf;
typedef __attribute__((ext_vector_type(8)))  float  v8f;

union FragB { uint4 q[2]; v16bf v; };

__device__ __forceinline__ unsigned short f2bfb(float f) {
  unsigned u = __float_as_uint(f);
  u += 0x7FFFu + ((u >> 16) & 1u);          // round-to-nearest-even
  return (unsigned short)(u >> 16);
}
__device__ __forceinline__ float bfb2f(unsigned short b) {
  return __uint_as_float(((unsigned)b) << 16);
}
__device__ __forceinline__ float gelu_tanh(float x) {
  const float c = 0.7978845608028654f;
  return 0.5f * x * (1.0f + tanhf(c * (x + 0.044715f * x * x * x)));
}

// CDNA5 async global->LDS DMA (ASYNCcnt-tracked, no VGPR round trip)
__device__ __forceinline__ void async_ld_b128(const void* gaddr, void* lgen) {
  unsigned l = (unsigned)(size_t)lgen;            // low 32 bits of the pointer give the LDS address
  asm volatile("global_load_async_to_lds_b128 %0, %1, off"
               :: "v"(l), "v"((unsigned long long)(size_t)gaddr)
               : "memory");
}
__device__ __forceinline__ void wait_async0() {
  asm volatile("s_wait_asynccnt 0x0" ::: "memory");
}

// ---------------- fp32 -> bf16 convert ----------------
__global__ void cvt_bf16_kernel(const float* __restrict__ s,
                                unsigned short* __restrict__ d, size_t n) {
  size_t i = (size_t)blockIdx.x * blockDim.x + threadIdx.x;
  size_t st = (size_t)gridDim.x * blockDim.x;
  for (; i < n; i += st) d[i] = f2bfb(s[i]);
}

// ---------------- router: softmax over E-1, top-2, normalized ----------------
__global__ void router_kernel(const float* __restrict__ xf,
                              const float* __restrict__ Wg,
                              int* __restrict__ flat_e,
                              float* __restrict__ flat_w) {
  int t = blockIdx.x * blockDim.x + threadIdx.x;
  if (t >= T_) return;
  float acc[7];
#pragma unroll
  for (int j = 0; j < 7; ++j) acc[j] = 0.0f;
  const float* xr = xf + (size_t)t * D_;
  for (int d = 0; d < D_; ++d) {
    float xv = xr[d];
#pragma unroll
    for (int j = 0; j < 7; ++j) acc[j] += xv * Wg[d * 7 + j];
  }
  float mx = acc[0];
#pragma unroll
  for (int j = 1; j < 7; ++j) mx = fmaxf(mx, acc[j]);
  float s = 0.0f;
#pragma unroll
  for (int j = 0; j < 7; ++j) { acc[j] = expf(acc[j] - mx); s += acc[j]; }
#pragma unroll
  for (int j = 0; j < 7; ++j) acc[j] /= s;
  int i1 = 0; float v1 = acc[0];
#pragma unroll
  for (int j = 1; j < 7; ++j) if (acc[j] > v1) { v1 = acc[j]; i1 = j; }
  int i2 = -1; float v2 = -1.0f;
#pragma unroll
  for (int j = 0; j < 7; ++j)
    if (j != i1 && acc[j] > v2) { v2 = acc[j]; i2 = j; }
  float sw = v1 + v2;
  flat_e[t * 2 + 0] = i1; flat_w[t * 2 + 0] = v1 / sw;
  flat_e[t * 2 + 1] = i2; flat_w[t * 2 + 1] = v2 / sw;
}

// ---------------- init maps ----------------
__global__ void init_kernel(int* __restrict__ tok_map, int* __restrict__ inv_slot) {
  int i = blockIdx.x * 256 + threadIdx.x;
  if (i < EG_) tok_map[i] = -1;
  if (i < T_ * TOPK_) inv_slot[i] = -1;
}

// ---------------- stable binning (matches stable argsort + capacity drop) ----
__global__ void __launch_bounds__(1024)
bin_kernel(const int* __restrict__ flat_e,
           int* __restrict__ tok_map, int* __restrict__ inv_slot) {
  const int e = blockIdx.x;            // expert
  const int tid = threadIdx.x;
  __shared__ int sc[1024];
  int base = 0;
  for (int c = 0; c < (T_ * TOPK_) / 1024; ++c) {
    int i = c * 1024 + tid;
    int m = (flat_e[i] == e) ? 1 : 0;
    sc[tid] = m;
    __syncthreads();
    for (int off = 1; off < 1024; off <<= 1) {
      int v = (tid >= off) ? sc[tid - off] : 0;
      __syncthreads();
      sc[tid] += v;
      __syncthreads();
    }
    int incl = sc[tid];
    int total = sc[1023];
    if (m) {
      int slot = base + incl - 1;
      if (slot < CAP_) {
        tok_map[e * CAP_ + slot] = i >> 1;     // source token id
        inv_slot[i] = e * CAP_ + slot;
      }
    }
    base += total;
    __syncthreads();
  }
}

// ---------------- gather cond rows into [EG, DC] bf16 ----------------
__global__ void gather_cond_kernel(const float* __restrict__ cf,
                                   const int* __restrict__ tok_map,
                                   unsigned short* __restrict__ cg) {
  int r = blockIdx.x;
  int tid = threadIdx.x;                 // 256 == DC_
  int tok = tok_map[r];
  float v = (tok >= 0) ? cf[(size_t)tok * DC_ + tid] : 0.0f;
  cg[(size_t)r * DC_ + tid] = f2bfb(v);
}

// ---------------- WMMA bf16 GEMM:  C[M,N] = A[M,K] * W[(e,)K,N] + bias ------
// 128x128 block tile, BK=64, double-buffered async-LDS pipeline,
// XOR-swizzled LDS to kill ds bank conflicts on frag loads.
#define BM 128
#define BN 128
#define BK 64

__global__ void __launch_bounds__(256)
gemm_bf16(const unsigned short* __restrict__ A,
          const unsigned short* __restrict__ W,
          const float* __restrict__ bias,
          void* __restrict__ Cout,
          int M, int N, int K, int rowsPerExpert, int mode)
{
  __shared__ __align__(16) unsigned short As[2][BM * BK];  // 2 x 16KB
  __shared__ __align__(16) unsigned short Bs[2][BK * BN];  // 2 x 16KB

  const int bm = blockIdx.y * BM;
  const int bn = blockIdx.x * BN;
  const int tid = threadIdx.x;
  const int lane = tid & 31;
  const int wave = tid >> 5;
  const int wr = wave & 3;               // 4 row-groups of 32
  const int wc = wave >> 2;              // 2 col-groups of 64

  const unsigned short* Wp = W;
  const float* bp = bias;
  if (rowsPerExpert > 0) {
    int e = bm / rowsPerExpert;
    Wp += (size_t)e * K * N;
    bp += (size_t)e * N;
  }

  v8f acc[2][4] = {};

  const int lr = lane & 15;
  const int kh = (lane >> 4) << 3;       // 0 or 8 (A K-half select)

  // LDS swizzles on 16B chunks:
  //   A rows are 128B (8 chunks):  chunk ^= (row>>1)&7
  //   B rows are 256B (16 chunks): chunk ^= row&15
  auto aswz = [](int row, int halfOff) {
    return row * BK + ((((halfOff >> 3) ^ ((row >> 1) & 7)) & 7) << 3);
  };
  auto bswz = [](int row, int halfOff) {
    return row * BN + ((((halfOff >> 3) ^ (row & 15)) & 15) << 3);
  };

  auto stage = [&](int buf, int k0) {
    // A tile: 128 x 64 halves = 1024 chunks -> 4 b128 per thread
#pragma unroll
    for (int it = 0; it < 4; ++it) {
      int q = tid + (it << 8);
      int r = q >> 3;
      int h = (q & 7) << 3;
      async_ld_b128(&A[(size_t)(bm + r) * K + (k0 + h)],
                    &As[buf][aswz(r, h)]);
    }
    // B tile: 64 x 128 halves = 1024 chunks -> 4 b128 per thread
#pragma unroll
    for (int it = 0; it < 4; ++it) {
      int q = tid + (it << 8);
      int r = q >> 4;
      int h = (q & 15) << 3;
      async_ld_b128(&Wp[(size_t)(k0 + r) * N + (bn + h)],
                    &Bs[buf][bswz(r, h)]);
    }
  };

  stage(0, 0);
  int buf = 0;
  for (int k0 = 0; k0 < K; k0 += BK) {
    wait_async0();          // current tile's DMA complete (this wave)
    __syncthreads();        // all waves' DMA complete; prev compute done
    if (k0 + BK < K) stage(buf ^ 1, k0 + BK);   // overlap DMA with compute

#pragma unroll
    for (int kk = 0; kk < BK; kk += 32) {
      FragB a[2], b[4];
#pragma unroll
      for (int mi = 0; mi < 2; ++mi) {
        int row = (wr << 5) + (mi << 4) + lr;
        a[mi].q[0] = *(const uint4*)&As[buf][aswz(row, kk + kh)];
        a[mi].q[1] = *(const uint4*)&As[buf][aswz(row, kk + kh + 16)];
      }
#pragma unroll
      for (int ni = 0; ni < 4; ++ni) {
        int col = (wc << 6) + (ni << 4) + lr;
        unsigned short* bu = (unsigned short*)&b[ni];
#pragma unroll
        for (int s = 0; s < 16; ++s) {
          int row = kk + kh + (s & 7) + ((s >> 3) << 4);
          bu[s] = Bs[buf][bswz(row, col) + (col & 7)];
        }
      }
#pragma unroll
      for (int mi = 0; mi < 2; ++mi)
#pragma unroll
        for (int ni = 0; ni < 4; ++ni)
          acc[mi][ni] = __builtin_amdgcn_wmma_f32_16x16x32_bf16(
              false, a[mi].v, false, b[ni].v, (short)0, acc[mi][ni],
              false, false);
    }
    buf ^= 1;
  }

  const int rowAdd = (lane >> 4) << 3;   // +8 rows for upper half-wave
#pragma unroll
  for (int mi = 0; mi < 2; ++mi) {
#pragma unroll
    for (int ni = 0; ni < 4; ++ni) {
      int col = bn + (wc << 6) + (ni << 4) + lr;
      float bv = bp ? bp[col] : 0.0f;
#pragma unroll
      for (int v = 0; v < 8; ++v) {
        int row = bm + (wr << 5) + (mi << 4) + v + rowAdd;
        float val = acc[mi][ni][v] + bv;
        if (mode == 1) val = gelu_tanh(val);
        if (mode == 2)
          ((float*)Cout)[(size_t)row * N + col] = val;
        else
          ((unsigned short*)Cout)[(size_t)row * N + col] = f2bfb(val);
      }
    }
  }
}

// ---------------- LayerNorm + adaLN modulation (per row) ----------------
__global__ void __launch_bounds__(256)
adaln_kernel(const float* __restrict__ xf,
             const int* __restrict__ tok_map,      // null => identity mapping
             const unsigned short* __restrict__ ada,
             unsigned short* __restrict__ hmod) {
  const int r = blockIdx.x;
  const int tid = threadIdx.x;                    // 256, 4 elems each
  const int tok = tok_map ? tok_map[r] : r;
  __shared__ float red[256];

  if (tok < 0) {
#pragma unroll
    for (int i = 0; i < 4; ++i) hmod[(size_t)r * D_ + i * 256 + tid] = 0;
    return;
  }
  float xv[4]; float s = 0.0f;
#pragma unroll
  for (int i = 0; i < 4; ++i) {
    xv[i] = xf[(size_t)tok * D_ + i * 256 + tid];
    s += xv[i];
  }
  red[tid] = s; __syncthreads();
  for (int off = 128; off > 0; off >>= 1) {
    if (tid < off) red[tid] += red[tid + off];
    __syncthreads();
  }
  float mu = red[0] / (float)D_;
  __syncthreads();
  float s2 = 0.0f;
#pragma unroll
  for (int i = 0; i < 4; ++i) { float d0 = xv[i] - mu; s2 += d0 * d0; }
  red[tid] = s2; __syncthreads();
  for (int off = 128; off > 0; off >>= 1) {
    if (tid < off) red[tid] += red[tid + off];
    __syncthreads();
  }
  float rstd = rsqrtf(red[0] / (float)D_ + 1e-5f);
#pragma unroll
  for (int i = 0; i < 4; ++i) {
    int j = i * 256 + tid;
    float h = (xv[i] - mu) * rstd;
    float shift = bfb2f(ada[(size_t)r * TD3_ + j]);
    float scale = bfb2f(ada[(size_t)r * TD3_ + D_ + j]);
    hmod[(size_t)r * D_ + j] = f2bfb(h * (1.0f + scale) + shift);
  }
}

// ---------------- final combine (deterministic, no atomics) ----------------
__global__ void __launch_bounds__(256)
combine_kernel(const float* __restrict__ ys,           // [T, D] shared MLP out
               const unsigned short* __restrict__ ada_s,
               const float* __restrict__ ye,           // [EG, D] routed MLP out
               const unsigned short* __restrict__ ada_e,
               const int* __restrict__ inv_slot,
               const float* __restrict__ flat_w,
               const float* __restrict__ mask,
               float* __restrict__ out) {
  const int t = blockIdx.x;
  const int tid = threadIdx.x;
  const int s0 = inv_slot[t * 2 + 0];
  const int s1 = inv_slot[t * 2 + 1];
  const float w0 = flat_w[t * 2 + 0];
  const float w1 = flat_w[t * 2 + 1];
  const float m = mask[t];
#pragma unroll
  for (int i = 0; i < 4; ++i) {
    int j = i * 256 + tid;
    float sh = ys[(size_t)t * D_ + j] *
               bfb2f(ada_s[(size_t)t * TD3_ + 2 * D_ + j]) * m;
    float moe = 0.0f;
    if (s0 >= 0)
      moe += ye[(size_t)s0 * D_ + j] *
             bfb2f(ada_e[(size_t)s0 * TD3_ + 2 * D_ + j]) * w0;
    if (s1 >= 0)
      moe += ye[(size_t)s1 * D_ + j] *
             bfb2f(ada_e[(size_t)s1 * TD3_ + 2 * D_ + j]) * w1;
    moe *= m;
    out[(size_t)t * D_ + j] = (sh + moe * (float)TOPK_) / (float)(TOPK_ + 1);
  }
}

// ---------------- host orchestration ----------------
extern "C" void kernel_launch(void* const* d_in, const int* in_sizes, int n_in,
                              void* d_out, int out_size, void* d_ws, size_t ws_size,
                              hipStream_t stream) {
  (void)in_sizes; (void)n_in; (void)out_size; (void)ws_size;
  const float* x      = (const float*)d_in[0];
  const float* cond   = (const float*)d_in[1];
  const float* mask   = (const float*)d_in[2];
  const float* Wg     = (const float*)d_in[3];
  const float* Wada_s = (const float*)d_in[4];
  const float* bada_s = (const float*)d_in[5];
  const float* W1_s   = (const float*)d_in[6];
  const float* b1_s   = (const float*)d_in[7];
  const float* W2_s   = (const float*)d_in[8];
  const float* b2_s   = (const float*)d_in[9];
  const float* Wada_e = (const float*)d_in[10];
  const float* bada_e = (const float*)d_in[11];
  const float* W1_e   = (const float*)d_in[12];
  const float* b1_e   = (const float*)d_in[13];
  const float* W2_e   = (const float*)d_in[14];
  const float* b2_e   = (const float*)d_in[15];
  float* out = (float*)d_out;

  char* ws = (char*)d_ws;
  size_t off = 0;
  auto carve = [&](size_t bytes) -> void* {
    void* p = ws + off;
    off = (off + bytes + 255) & ~(size_t)255;
    return p;
  };
  unsigned short* xf_bf     = (unsigned short*)carve((size_t)T_ * D_ * 2);
  unsigned short* cf_bf     = (unsigned short*)carve((size_t)T_ * DC_ * 2);
  unsigned short* cg_bf     = (unsigned short*)carve((size_t)EG_ * DC_ * 2);
  unsigned short* Wada_s_bf = (unsigned short*)carve((size_t)DC_ * TD3_ * 2);
  unsigned short* W1_s_bf   = (unsigned short*)carve((size_t)D_ * HID_ * 2);
  unsigned short* W2_s_bf   = (unsigned short*)carve((size_t)HID_ * D_ * 2);
  unsigned short* Wada_e_bf = (unsigned short*)carve((size_t)E_ * DC_ * TD3_ * 2);
  unsigned short* W1_e_bf   = (unsigned short*)carve((size_t)E_ * D_ * HID_ * 2);
  unsigned short* W2_e_bf   = (unsigned short*)carve((size_t)E_ * HID_ * D_ * 2);
  int*            flat_e    = (int*)carve((size_t)T_ * TOPK_ * 4);
  float*          flat_w    = (float*)carve((size_t)T_ * TOPK_ * 4);
  int*            tok_map   = (int*)carve((size_t)EG_ * 4);
  int*            inv_slot  = (int*)carve((size_t)T_ * TOPK_ * 4);
  unsigned short* ada_s_bf  = (unsigned short*)carve((size_t)T_ * TD3_ * 2);
  unsigned short* ada_e_bf  = (unsigned short*)carve((size_t)EG_ * TD3_ * 2);
  unsigned short* hmod_s    = (unsigned short*)carve((size_t)T_ * D_ * 2);
  unsigned short* hmod_e    = (unsigned short*)carve((size_t)EG_ * D_ * 2);
  unsigned short* h1_s      = (unsigned short*)carve((size_t)T_ * HID_ * 2);
  unsigned short* h1_e      = (unsigned short*)carve((size_t)EG_ * HID_ * 2);
  float*          ys        = (float*)carve((size_t)T_ * D_ * 4);
  float*          ye        = (float*)carve((size_t)EG_ * D_ * 4);

  auto cvt = [&](const float* s, unsigned short* d, size_t n) {
    cvt_bf16_kernel<<<2048, 256, 0, stream>>>(s, d, n);
  };
  // fp32 -> bf16 for WMMA inputs
  cvt(x,      xf_bf,     (size_t)T_ * D_);
  cvt(cond,   cf_bf,     (size_t)T_ * DC_);
  cvt(Wada_s, Wada_s_bf, (size_t)DC_ * TD3_);
  cvt(W1_s,   W1_s_bf,   (size_t)D_ * HID_);
  cvt(W2_s,   W2_s_bf,   (size_t)HID_ * D_);
  cvt(Wada_e, Wada_e_bf, (size_t)E_ * DC_ * TD3_);
  cvt(W1_e,   W1_e_bf,   (size_t)E_ * D_ * HID_);
  cvt(W2_e,   W2_e_bf,   (size_t)E_ * HID_ * D_);

  // routing + deterministic stable binning
  router_kernel<<<T_ / 256, 256, 0, stream>>>(x, Wg, flat_e, flat_w);
  init_kernel<<<(EG_ + 255) / 256, 256, 0, stream>>>(tok_map, inv_slot);
  bin_kernel<<<E_, 1024, 0, stream>>>(flat_e, tok_map, inv_slot);
  gather_cond_kernel<<<EG_, 256, 0, stream>>>(cond, tok_map, cg_bf);

  // ada = cond @ Wada + bada   (routed + shared)
  {
    dim3 g(TD3_ / BN, EG_ / BM);
    gemm_bf16<<<g, 256, 0, stream>>>(cg_bf, Wada_e_bf, bada_e, ada_e_bf,
                                     EG_, TD3_, DC_, CAP_, 0);
  }
  {
    dim3 g(TD3_ / BN, T_ / BM);
    gemm_bf16<<<g, 256, 0, stream>>>(cf_bf, Wada_s_bf, bada_s, ada_s_bf,
                                     T_, TD3_, DC_, 0, 0);
  }

  // LayerNorm + modulation
  adaln_kernel<<<EG_, 256, 0, stream>>>(x, tok_map, ada_e_bf, hmod_e);
  adaln_kernel<<<T_, 256, 0, stream>>>(x, nullptr, ada_s_bf, hmod_s);

  // h1 = gelu(hmod @ W1 + b1)
  {
    dim3 g(HID_ / BN, EG_ / BM);
    gemm_bf16<<<g, 256, 0, stream>>>(hmod_e, W1_e_bf, b1_e, h1_e,
                                     EG_, HID_, D_, CAP_, 1);
  }
  {
    dim3 g(HID_ / BN, T_ / BM);
    gemm_bf16<<<g, 256, 0, stream>>>(hmod_s, W1_s_bf, b1_s, h1_s,
                                     T_, HID_, D_, 0, 1);
  }

  // y = h1 @ W2 + b2  (fp32 out)
  {
    dim3 g(D_ / BN, EG_ / BM);
    gemm_bf16<<<g, 256, 0, stream>>>(h1_e, W2_e_bf, b2_e, ye,
                                     EG_, D_, HID_, CAP_, 2);
  }
  {
    dim3 g(D_ / BN, T_ / BM);
    gemm_bf16<<<g, 256, 0, stream>>>(h1_s, W2_s_bf, b2_s, ys,
                                     T_, D_, HID_, 0, 2);
  }

  // gate * h * mask, router weights, (shared + 2*moe)/3
  combine_kernel<<<T_, 256, 0, stream>>>(ys, ada_s_bf, ye, ada_e_bf,
                                         inv_slot, flat_w, mask, out);
}
